// TemporalIntegrator_76141180224160
// MI455X (gfx1250) — matrix operand order repaired and checked by reading, not compiled, not run on hardware
//
#include <hip/hip_runtime.h>
#include <hip/hip_fp16.h>
#include <math.h>

typedef __attribute__((ext_vector_type(16))) _Float16 v16h;
typedef __attribute__((ext_vector_type(8)))  float    v8f;

typedef unsigned int tdm_v4u __attribute__((ext_vector_type(4)));
typedef int          tdm_v4i __attribute__((ext_vector_type(4)));
typedef int          tdm_v8i __attribute__((ext_vector_type(8)));

#define NFREQ 10
#define CFEAT 128
#define KPAD  352          // 340 padded to 11 * 32
#define KT    11           // K tiles of 32
#define MT    8            // 128 / 16 output-channel tiles

// ---------------------------------------------------------------------------
// Kernel 1: 2x downsample (align_corners=False bilinear at exact 2x == 2x2 box
// average) followed by *0.5 flow rescale  ->  out = 0.125 * (4-tap sum)
// ---------------------------------------------------------------------------
__global__ void downsample_flow_k(const float* __restrict__ in,
                                  float* __restrict__ out,
                                  int C, int Hi) {
  int Ho = Hi >> 1;
  int idx = blockIdx.x * blockDim.x + threadIdx.x;
  int total = C * Ho * Ho;
  if (idx >= total) return;
  int x = idx % Ho;
  int y = (idx / Ho) % Ho;
  int c = idx / (Ho * Ho);
  const float* p = in + (size_t)c * Hi * Hi + (size_t)(2 * y) * Hi + 2 * x;
  out[idx] = 0.125f * (p[0] + p[1] + p[Hi] + p[Hi + 1]);
}

// ---------------------------------------------------------------------------
// Kernel 2: pack f32 weights [128 x 340] into f16 WMMA A-fragments,
// K zero-padded to 352.  Layout: ((mt*11 + kt)*32 + lane)*16 + elem, where
// (16-bit A 16x32, this code's own k order, shared with B):
//   lane half = lane>>4, M = mt*16 + (lane&15)
//   elem e -> K = kt*32 + half*16 + e  (contiguous)
// ---------------------------------------------------------------------------
__global__ void pack_w_k(const float* __restrict__ W, _Float16* __restrict__ wp) {
  int idx = blockIdx.x * blockDim.x + threadIdx.x;   // MT*KT*32*16 = 45056
  if (idx >= MT * KT * 32 * 16) return;
  int e    = idx & 15;
  int rest = idx >> 4;
  int lane = rest & 31;
  int rest2 = rest >> 5;
  int kt = rest2 % KT;
  int mt = rest2 / KT;
  int half = lane >> 4;
  int M = mt * 16 + (lane & 15);
  int K = kt * 32 + half * 16 + e;
  float v = (K < 340) ? W[(size_t)M * 340 + K] : 0.0f;
  wp[idx] = (_Float16)v;
}

// ---------------------------------------------------------------------------
// Kernel 3: build fused activations (352 f16 channels per pixel, pixel-major)
// ---------------------------------------------------------------------------
struct Taps { int l00, l01, l10, l11; float w00, w01, w10, w11; };

__device__ inline Taps mk_taps(float px, float py, int H, int W) {
  float x0f = floorf(px), y0f = floorf(py);
  float ax = px - x0f, ay = py - y0f;
  int x0 = (int)x0f, y0 = (int)y0f;
  int x1 = x0 + 1,   y1 = y0 + 1;
  float v00 = (x0 >= 0 && x0 < W && y0 >= 0 && y0 < H) ? 1.f : 0.f;
  float v01 = (x1 >= 0 && x1 < W && y0 >= 0 && y0 < H) ? 1.f : 0.f;
  float v10 = (x0 >= 0 && x0 < W && y1 >= 0 && y1 < H) ? 1.f : 0.f;
  float v11 = (x1 >= 0 && x1 < W && y1 >= 0 && y1 < H) ? 1.f : 0.f;
  int cx0 = x0 < 0 ? 0 : (x0 > W - 1 ? W - 1 : x0);
  int cx1 = x1 < 0 ? 0 : (x1 > W - 1 ? W - 1 : x1);
  int cy0 = y0 < 0 ? 0 : (y0 > H - 1 ? H - 1 : y0);
  int cy1 = y1 < 0 ? 0 : (y1 > H - 1 ? H - 1 : y1);
  Taps t;
  t.w00 = (1.f - ax) * (1.f - ay) * v00;
  t.w01 = ax * (1.f - ay) * v01;
  t.w10 = (1.f - ax) * ay * v10;
  t.w11 = ax * ay * v11;
  t.l00 = cy0 * W + cx0;
  t.l01 = cy0 * W + cx1;
  t.l10 = cy1 * W + cx0;
  t.l11 = cy1 * W + cx1;
  float ms = t.w00 + t.w01 + t.w10 + t.w11;
  if (ms < 0.9999f) { t.w00 = t.w01 = t.w10 = t.w11 = 0.f; }  // PWC mask
  return t;
}

__global__ void prep_fused_k(const float* __restrict__ fN1,
                             const float* __restrict__ f0t,
                             const float* __restrict__ fP1,
                             const float* __restrict__ flowN1,   // [4,2,384,384]
                             const float* __restrict__ flowP1,
                             const float* __restrict__ mask384,  // [4,1,384,384]
                             const float* __restrict__ pyrN,     // [4,2,H,W]
                             const float* __restrict__ pyrP,
                             _Float16* __restrict__ fused,
                             int H) {
  const int FS = 384;
  int W = H;
  int idx = blockIdx.x * blockDim.x + threadIdx.x;
  int total = 4 * H * W;
  if (idx >= total) return;
  int x = idx % W;
  int y = (idx / W) % H;
  int b = idx / (W * H);
  int HW = H * W;

  // bilinear coords into 384 grid (align_corners=False, clipped)
  float sxf = (x + 0.5f) * (384.0f / (float)W) - 0.5f;
  float syf = (y + 0.5f) * (384.0f / (float)H) - 0.5f;
  sxf = fminf(fmaxf(sxf, 0.f), 383.f);
  syf = fminf(fmaxf(syf, 0.f), 383.f);
  int rx0 = (int)floorf(sxf); int rx1 = rx0 + 1 > 383 ? 383 : rx0 + 1;
  int ry0 = (int)floorf(syf); int ry1 = ry0 + 1 > 383 ? 383 : ry0 + 1;
  float rwx = sxf - (float)rx0, rwy = syf - (float)ry0;

  auto samp384 = [&](const float* t, int c, int Cn) -> float {
    const float* p = t + ((size_t)(b * Cn + c)) * FS * FS;
    float a0 = p[ry0 * FS + rx0] * (1.f - rwx) + p[ry0 * FS + rx1] * rwx;
    float a1 = p[ry1 * FS + rx0] * (1.f - rwx) + p[ry1 * FS + rx1] * rwx;
    return a0 * (1.f - rwy) + a1 * rwy;
  };

  float scaleF = (float)H / 384.0f;
  float wfxN = samp384(flowN1, 0, 2) * scaleF;
  float wfyN = samp384(flowN1, 1, 2) * scaleF;
  float wfxP = samp384(flowP1, 0, 2) * scaleF;
  float wfyP = samp384(flowP1, 1, 2) * scaleF;
  float m  = samp384(mask384, 0, 1);
  float m1 = 1.f - m;

  Taps tN = mk_taps((float)x + wfxN, (float)y + wfyN, H, W);
  Taps tP = mk_taps((float)x + wfxP, (float)y + wfyP, H, W);

  _Float16* fp = fused + (size_t)idx * KPAD;
  int yx = y * W + x;

  // e0: center features
  for (int c = 0; c < CFEAT; c++)
    fp[c] = (_Float16)f0t[((size_t)(b * CFEAT + c)) * HW + yx];

  // proj(0): [0,0] then per freq [sin=0,0, cos=1,1]
  fp[128] = (_Float16)0.f; fp[129] = (_Float16)0.f;
  for (int k = 0; k < NFREQ; k++) {
    fp[130 + 4 * k] = (_Float16)0.f;
    fp[131 + 4 * k] = (_Float16)0.f;
    fp[132 + 4 * k] = (_Float16)1.f;
    fp[133 + 4 * k] = (_Float16)1.f;
  }

  // blended warped features
  for (int c = 0; c < CFEAT; c++) {
    size_t bc = ((size_t)(b * CFEAT + c)) * HW;
    float vn = tN.w00 * fN1[bc + tN.l00] + tN.w01 * fN1[bc + tN.l01] +
               tN.w10 * fN1[bc + tN.l10] + tN.w11 * fN1[bc + tN.l11];
    float vp = tP.w00 * fP1[bc + tP.l00] + tP.w01 * fP1[bc + tP.l01] +
               tP.w10 * fP1[bc + tP.l10] + tP.w11 * fP1[bc + tP.l11];
    fp[170 + c] = (_Float16)(m * vn + m1 * vp);
  }

  // blended sin/cos projections of pyramid flows
  size_t pb = ((size_t)(b * 2)) * HW + yx;
  float pxN = pyrN[pb], pyN = pyrN[pb + HW];
  float pxP = pyrP[pb], pyP = pyrP[pb + HW];
  fp[298] = (_Float16)(m * pxN + m1 * pxP);
  fp[299] = (_Float16)(m * pyN + m1 * pyP);
  float fr = 1.0f;
  for (int k = 0; k < NFREQ; k++) {
    float snx = __sinf(pxN * fr), sny = __sinf(pyN * fr);
    float cnx = __cosf(pxN * fr), cny = __cosf(pyN * fr);
    float spx = __sinf(pxP * fr), spy = __sinf(pyP * fr);
    float cpx = __cosf(pxP * fr), cpy = __cosf(pyP * fr);
    fp[300 + 4 * k] = (_Float16)(m * snx + m1 * spx);
    fp[301 + 4 * k] = (_Float16)(m * sny + m1 * spy);
    fp[302 + 4 * k] = (_Float16)(m * cnx + m1 * cpx);
    fp[303 + 4 * k] = (_Float16)(m * cny + m1 * cpy);
    fr *= 2.0f;
  }
  for (int c = 340; c < KPAD; c++) fp[c] = (_Float16)0.f;
}

// ---------------------------------------------------------------------------
// Kernel 4: WMMA GEMM  y[128, pix] = W[128,352] * fused[352, pix], bias + ELU
// 8 waves/block, one 16-pixel tile per wave.  Per mt-step, the mt's 11 A-tile
// set (11,264 B) is staged global->LDS once per block: by the Tensor Data
// Mover (1-D D#, data_size=2B, tile_dim0=5632) when the builtin exists,
// else by a cooperative copy.  A-fragments then come from ds_load_b128.
// B frag (32x16, same k order as the A pack): lane -> N = lane&15 (pixel),
//   K half = lane>>4, elem e -> K = kt*32 + (lane>>4)*16 + e  (contiguous)
// ---------------------------------------------------------------------------
__global__ void gemm_k(const _Float16* __restrict__ fused,   // this scale's base
                       const _Float16* __restrict__ wpack,   // MT*KT*32*16
                       const float* __restrict__ bias,       // 128
                       float* __restrict__ out,              // this scale's base
                       int HW) {
  __shared__ __align__(16) _Float16 sW[KT * 32 * 16];        // 11,264 bytes
  int tid  = threadIdx.x;
  int wave = tid >> 5;
  int lane = tid & 31;
  int wid  = blockIdx.x * 8 + wave;                          // grid is exact

  int p = wid * 16 + (lane & 15);           // pixel index (b*HW + q)
  int baseK = (lane >> 4) << 4;             // 0 or 16
  const v16h* bp = (const v16h*)(fused + (size_t)p * KPAD + baseK);

  v16h bfrag[KT];
#pragma unroll
  for (int k = 0; k < KT; k++) bfrag[k] = bp[2 * k];   // +32 halfs per K tile

  int halfsel = lane >> 4;
  int b = p / HW;
  int q = p - b * HW;

  for (int mt = 0; mt < MT; mt++) {
    __syncthreads();                                   // prev sW fully consumed
    const _Float16* src = wpack + (size_t)mt * KT * 32 * 16;
#if defined(__gfx1250__) && __has_builtin(__builtin_amdgcn_tensor_load_to_lds)
    if (wave == 0) {                                   // wave-uniform guard
      unsigned long long ga = (unsigned long long)(size_t)src;
      unsigned int lds = (unsigned int)(size_t)&sW[0]; // addr[31:0] = LDS offset
      const unsigned int NE = KT * 32 * 16;            // 5632 16-bit elements
      // D# group0: count=1 | lds_addr | global_addr[56:0] | type=2
      tdm_v4u g0 = { 1u,
                     lds,
                     (unsigned int)(ga & 0xFFFFFFFFu),
                     (unsigned int)(((ga >> 32) & 0x1FFFFFFu) | (2u << 30)) };
      // D# group1: data_size=2B; tensor_dim0=NE; tensor_dim1=1; tile_dim0=NE;
      //            tile_dim1=1; tensor_dim0_stride=NE
      tdm_v8i g1 = { (int)(1u << 16),
                     (int)((NE & 0xFFFFu) << 16),
                     (int)((NE >> 16) | (1u << 16)),
                     (int)(NE << 16),
                     1,
                     (int)NE,
                     0, 0 };
      tdm_v4i z4 = {0, 0, 0, 0};
#if __clang_major__ >= 23
      tdm_v8i z8 = {0, 0, 0, 0, 0, 0, 0, 0};
      __builtin_amdgcn_tensor_load_to_lds(g0, g1, z4, z4, z8, 0);
#else
      __builtin_amdgcn_tensor_load_to_lds(g0, g1, z4, z4, 0);
#endif
      __builtin_amdgcn_s_wait_tensorcnt((short)0);
    }
#else
    for (int i = tid; i < (KT * 32 * 16) / 8; i += 256)
      ((float4*)sW)[i] = ((const float4*)src)[i];
#endif
    __syncthreads();                                   // sW ready for all waves

    v8f acc = {};
#pragma unroll
    for (int k = 0; k < KT; k++) {
      v16h a = *(const v16h*)&sW[((unsigned)k * 32 + lane) * 16];
      acc = __builtin_amdgcn_wmma_f32_16x16x32_f16(
          /*neg_a=*/false, a, /*neg_b=*/false, bfrag[k],
          /*c_mod=*/(short)0, acc, /*reuse_a=*/false, /*reuse_b=*/false);
    }
#pragma unroll
    for (int v = 0; v < 8; v++) {
      int mrow = mt * 16 + halfsel * 8 + v;
      float val = acc[v] + bias[mrow];
      val = val > 0.f ? val : (__expf(val) - 1.0f);    // ELU
      out[((size_t)(b * CFEAT + mrow)) * HW + q] = val;
    }
  }
}

// ---------------------------------------------------------------------------
extern "C" void kernel_launch(void* const* d_in, const int* in_sizes, int n_in,
                              void* d_out, int out_size, void* d_ws, size_t ws_size,
                              hipStream_t stream) {
  const float* fN1[4] = {(const float*)d_in[0], (const float*)d_in[1],
                         (const float*)d_in[2], (const float*)d_in[3]};
  const float* f0v[4] = {(const float*)d_in[4], (const float*)d_in[5],
                         (const float*)d_in[6], (const float*)d_in[7]};
  const float* fP1[4] = {(const float*)d_in[8], (const float*)d_in[9],
                         (const float*)d_in[10], (const float*)d_in[11]};
  const float* flowN1 = (const float*)d_in[12];
  const float* flowP1 = (const float*)d_in[13];
  const float* mask   = (const float*)d_in[14];
  const float* Wv[4]  = {(const float*)d_in[15], (const float*)d_in[17],
                         (const float*)d_in[19], (const float*)d_in[21]};
  const float* bv[4]  = {(const float*)d_in[16], (const float*)d_in[18],
                         (const float*)d_in[20], (const float*)d_in[22]};
  float* out = (float*)d_out;

  const int    sizes[4]   = {192, 96, 48, 24};
  const size_t pxbase[4]  = {0, 147456, 184320, 193536};          // pixels
  const size_t outbase[4] = {0, 18874368, 23592960, 24772608};    // f32 elems
  const size_t pyrOff[4]  = {0, 294912, 368640, 387072};          // f32 elems
  const size_t NPIX = 195840;
  const size_t PYRF = 391680;

  char* ws = (char*)d_ws;
  size_t off = 0;
  _Float16* fused = (_Float16*)(ws + off); off += NPIX * KPAD * sizeof(_Float16);
  float* pyrN1 = (float*)(ws + off); off += PYRF * sizeof(float);
  float* pyrP1 = (float*)(ws + off); off += PYRF * sizeof(float);
  _Float16* wpack = (_Float16*)(ws + off); off += (size_t)4 * MT * KT * 32 * 16 * sizeof(_Float16);
  (void)ws_size; (void)in_sizes; (void)n_in; (void)out_size;

  // 1) flow pyramids (iterated 2x downsample with *0.5 flow rescale)
  const float* srcN = flowN1;
  const float* srcP = flowP1;
  int Hi = 384;
  for (int l = 0; l < 4; l++) {
    int Ho = Hi >> 1;
    int total = 8 * Ho * Ho;   // B*2 channels
    downsample_flow_k<<<(total + 255) / 256, 256, 0, stream>>>(srcN, pyrN1 + pyrOff[l], 8, Hi);
    downsample_flow_k<<<(total + 255) / 256, 256, 0, stream>>>(srcP, pyrP1 + pyrOff[l], 8, Hi);
    srcN = pyrN1 + pyrOff[l];
    srcP = pyrP1 + pyrOff[l];
    Hi = Ho;
  }

  // 2) pack weights into WMMA A-fragment order
  for (int s = 0; s < 4; s++) {
    int total = MT * KT * 32 * 16;
    pack_w_k<<<(total + 255) / 256, 256, 0, stream>>>(Wv[s], wpack + (size_t)s * total);
  }

  // 3) fused activation build
  for (int s = 0; s < 4; s++) {
    int H = sizes[s];
    int total = 4 * H * H;
    prep_fused_k<<<(total + 255) / 256, 256, 0, stream>>>(
        fN1[s], f0v[s], fP1[s], flowN1, flowP1, mask,
        pyrN1 + pyrOff[s], pyrP1 + pyrOff[s],
        fused + pxbase[s] * KPAD, H);
  }

  // 4) WMMA GEMM + bias + ELU (tiles per scale are multiples of 8 -> exact grid)
  for (int s = 0; s < 4; s++) {
    int H = sizes[s];
    int HW = H * H;
    int ntiles = 4 * HW / 16;
    int blocks = ntiles / 8;                  // 8 waves (tiles) per block
    gemm_k<<<blocks, 256, 0, stream>>>(
        fused + pxbase[s] * KPAD,
        wpack + (size_t)s * MT * KT * 32 * 16,
        bv[s], out + outbase[s], HW);
  }
}
